// DCNv3_2061584302095
// MI455X (gfx1250) — compile-verified, matrix operand order checked by reading, not run on hardware
//
#include <hip/hip_runtime.h>
#include <hip/hip_bf16.h>

// ---------------------------------------------------------------------------
// DCNv3 forward for MI455X (gfx1250, wave32).
//   B=4, C=128, H=W=128, G=4, P=9, gc=32. Channels-last reinterpretation
//   makes input/output layouts flat-compatible with (B,C,H,W) buffers.
// Pipeline:
//   1) dcn_prep  : pack [W_off | W_mask | 0-pad] into WMMA *fragment order*
//                  f16 B-matrix ([ntile][kstep][lane][16]) + f32 bias
//   2) dcn_gemm  : offset/mask GEMM (65536 x 128) x (128 x 112) via
//                  v_wmma_f32_16x16x32_f16. 32-row M tile per block, B
//                  fragments register-resident, 8 unrolled WMMAs per wave.
//   3) dcn_sample: per-(pixel,group) wave, lane = channel; softmax + 9
//                  bilinear taps, coalesced 128B channel reads (L2 resident)
// ---------------------------------------------------------------------------

typedef __attribute__((ext_vector_type(16))) _Float16 v16h;
typedef __attribute__((ext_vector_type(8)))  float    v8f;

#define CB   128          // channels
#define HH   128
#define WW   128
#define GG   4
#define PP   9
#define GC   32           // channels per group
#define MROWS 65536       // B*H*W
#define NOFF 72           // G*P*2
#define NMASK 36          // G*P
#define NPAD 112          // 72+36 padded to 7x16
#define NTILES 7
#define KSTEPS 4          // K=128 in 4 steps of 32
#define MTILE 32          // rows per block (2 WMMA sub-tiles)

// ws layout (bytes)
#define WS_WC    0        // _Float16[7*4*32*16] = 28672 B (fragment order)
#define WS_BIAS  28672    // float[112]          = 448 B
#define WS_OM    32768    // float[65536*112]    = 29,360,128 B

// K index inside a 32-wide step for fragment half j, half-select hi
// (CDNA5 ISA 7.12.2: 16-bit A 16x32 layout; B mirrored with N instead of M)
__device__ __forceinline__ int frag_k(int j, int hi) {
  const int v = j >> 1, pbit = j & 1;
  const int kloc = (v < 4) ? (2 * v + pbit) : (16 + 2 * (v - 4) + pbit);
  return kloc + hi * 8;
}

// ---------------------------------------------------------------------------
// Kernel 1: build B in fragment order + bias.
//   WcF[((ntile*4 + kk)*32 + lane)*16 + j] = Wc[k][n],
//   k = kk*32 + frag_k(j, lane>>4),  n = ntile*16 + (lane&15)
// ---------------------------------------------------------------------------
__global__ __launch_bounds__(256)
void dcn_prep(const float* __restrict__ Woff, const float* __restrict__ boff,
              const float* __restrict__ Wmask, const float* __restrict__ bmask,
              _Float16* __restrict__ WcF, float* __restrict__ bias) {
  int i = blockIdx.x * 256 + threadIdx.x;
  if (i < NTILES * KSTEPS * 32 * 16) {
    const int j     = i & 15;
    const int lane  = (i >> 4) & 31;
    const int kk    = (i >> 9) & 3;
    const int ntile = i >> 11;
    const int k = kk * 32 + frag_k(j, lane >> 4);
    const int n = ntile * 16 + (lane & 15);
    float v = 0.0f;
    if (n < NOFF)               v = Woff[k * NOFF + n];
    else if (n < NOFF + NMASK)  v = Wmask[k * NMASK + (n - NOFF)];
    WcF[i] = (_Float16)v;
  }
  if (i < NPAD) {
    float v = 0.0f;
    if (i < NOFF)               v = boff[i];
    else if (i < NOFF + NMASK)  v = bmask[i - NOFF];
    bias[i] = v;
  }
}

// ---------------------------------------------------------------------------
// Kernel 2: offset/mask GEMM. 7 waves per block (one N-tile each), 32 M rows
// per block. B fragments loaded once as contiguous 32B vectors; 8 WMMAs.
// ---------------------------------------------------------------------------
__global__ __launch_bounds__(224)
void dcn_gemm(const float* __restrict__ x, const _Float16* __restrict__ WcF,
              const float* __restrict__ bias, float* __restrict__ out) {
  __shared__ _Float16 As[MTILE * CB];       // 8 KB A tile, f16

  const int tid = threadIdx.x;
  const long rowbase = (long)blockIdx.x * MTILE;

  // cooperative load+convert: 32 rows x 128 ch, float2 -> 2x f16
  for (int i = tid; i < (MTILE * CB) / 2; i += 224) {
    const int idx = i * 2;
    const int r = idx >> 7, c = idx & 127;
    const float2 f = *(const float2*)&x[(rowbase + r) * CB + c];
    As[idx]     = (_Float16)f.x;
    As[idx + 1] = (_Float16)f.y;
  }

  const int ntile = tid >> 5;               // wave id = N tile (0..6)
  const int lane  = tid & 31;
  const int mn    = lane & 15;              // row (A) / col (B,C,D)
  const int hi    = lane >> 4;              // +8 on K (A/B), +8 on M (C/D)

  // B fragments: register resident, 32B contiguous per lane per k-step
  v16h bfrag[KSTEPS];
#pragma unroll
  for (int kk = 0; kk < KSTEPS; ++kk) {
    bfrag[kk] = *(const v16h*)&WcF[(((ntile * KSTEPS) + kk) * 32 + lane) * 16];
  }

  __syncthreads();

  v8f acc[2] = {{}, {}};
#pragma unroll
  for (int rb = 0; rb < 2; ++rb) {
    const int arow = rb * 16 + mn;
#pragma unroll
    for (int kk = 0; kk < KSTEPS; ++kk) {
      v16h a;
#pragma unroll
      for (int j = 0; j < 16; ++j) {
        a[j] = As[arow * CB + kk * 32 + frag_k(j, hi)];
      }
      acc[rb] = __builtin_amdgcn_wmma_f32_16x16x32_f16(
          false, a, false, bfrag[kk], (short)0, acc[rb], false, false);
    }
  }

  const int ncol = ntile * 16 + mn;
  const float bv = bias[ncol];
#pragma unroll
  for (int rb = 0; rb < 2; ++rb) {
#pragma unroll
    for (int r = 0; r < 8; ++r) {
      const int mrow = rb * 16 + r + hi * 8;       // C/D VGPR layout
      out[(rowbase + mrow) * NPAD + ncol] = acc[rb][r] + bv;
    }
  }
}

// ---------------------------------------------------------------------------
// Kernel 3: softmax + bilinear sampling. One wave per (pixel, group),
// lane = channel within group -> every tap is one coalesced 128B line.
// ---------------------------------------------------------------------------
__device__ __forceinline__
float dcn_tap(const float* __restrict__ xb, int ix, int iy, float wgt) {
  // padded coords [0,130); pad ring (border width 1) holds zeros.
  if (ix < 1 || ix > WW || iy < 1 || iy > HH) return 0.0f;   // covers OOB + pad
  return wgt * xb[((long)(iy - 1) * WW + (ix - 1)) * CB];
}

__global__ __launch_bounds__(256)
void dcn_sample(const float* __restrict__ x, const float* __restrict__ om,
                float* __restrict__ out) {
  const int lane = threadIdx.x & 31;
  const int wid  = blockIdx.x * 8 + (threadIdx.x >> 5);  // (pixel,group) id
  const int g = wid & 3;
  const int r = wid >> 2;                                 // pixel row 0..65535
  const int b = r >> 14;
  const int h = (r >> 7) & 127;
  const int w = r & 127;

  const float* __restrict__ orow = om + (long)r * NPAD;

  // softmax over 9 mask logits for this group (uniform across lanes)
  float lg[PP];
  float mx = -1e30f;
#pragma unroll
  for (int p = 0; p < PP; ++p) {
    lg[p] = orow[NOFF + g * PP + p];
    mx = fmaxf(mx, lg[p]);
  }
  float s = 0.0f;
#pragma unroll
  for (int p = 0; p < PP; ++p) { lg[p] = __expf(lg[p] - mx); s += lg[p]; }
  const float inv = 1.0f / s;

  // base pointer: batch + group-channel + lane channel
  const float* __restrict__ xb = x + ((long)b * HH * WW) * CB + g * GC + lane;

  float acc = 0.0f;
#pragma unroll
  for (int p = 0; p < PP; ++p) {
    const float dx = orow[g * (PP * 2) + 2 * p];
    const float dy = orow[g * (PP * 2) + 2 * p + 1];
    // px = loc_x*W_in - 0.5 simplifies to (w+1) + kx + dx  (kx,ky in {-1,0,1})
    const float px = (float)(w + 1 + (p / 3) - 1) + dx;
    const float py = (float)(h + 1 + (p % 3) - 1) + dy;
    const float x0 = floorf(px), y0 = floorf(py);
    const float tx = px - x0,   ty = py - y0;
    const int ix0 = (int)x0, iy0 = (int)y0;
    const float mwt = lg[p] * inv;

    float v = dcn_tap(xb, ix0,     iy0,     (1.0f - tx) * (1.0f - ty))
            + dcn_tap(xb, ix0 + 1, iy0,     tx * (1.0f - ty))
            + dcn_tap(xb, ix0,     iy0 + 1, (1.0f - tx) * ty)
            + dcn_tap(xb, ix0 + 1, iy0 + 1, tx * ty);
    acc += mwt * v;
  }

  // output is channels-last flat == (B,C,H,W) reinterpretation
  out[(long)r * CB + g * GC + lane] = acc;
}

// ---------------------------------------------------------------------------
extern "C" void kernel_launch(void* const* d_in, const int* in_sizes, int n_in,
                              void* d_out, int out_size, void* d_ws, size_t ws_size,
                              hipStream_t stream) {
  const float* inp   = (const float*)d_in[0];
  const float* Woff  = (const float*)d_in[1];
  const float* boff  = (const float*)d_in[2];
  const float* Wmask = (const float*)d_in[3];
  const float* bmask = (const float*)d_in[4];

  char* ws = (char*)d_ws;
  _Float16* WcF = (_Float16*)(ws + WS_WC);
  float*   bias = (float*)(ws + WS_BIAS);
  float*   om   = (float*)(ws + WS_OM);

  dcn_prep<<<(NTILES * KSTEPS * 32 * 16 + 255) / 256, 256, 0, stream>>>(
      Woff, boff, Wmask, bmask, WcF, bias);
  dcn_gemm<<<MROWS / MTILE, 224, 0, stream>>>(inp, WcF, bias, om);
  dcn_sample<<<(MROWS * GG) / 8, 256, 0, stream>>>(inp, om, (float*)d_out);
}
